// LinearAttention_80917183856930
// MI455X (gfx1250) — matrix operand order repaired and checked by reading, not compiled, not run on hardware
//
#include <hip/hip_runtime.h>
#include <hip/hip_bf16.h>
#include <stdint.h>

typedef __bf16 bf16;
typedef __attribute__((ext_vector_type(16))) __bf16 bf16x16;
typedef __attribute__((ext_vector_type(8)))  float  f32x8;

#define B_      4
#define SEQ     4096
#define CDIM    512
#define H_      8
#define HD      64
#define NTOK    (B_ * SEQ)          // 16384
#define BH      (B_ * H_)           // 32

static __device__ __forceinline__ f32x8 wmma_bf(bf16x16 a, bf16x16 b, f32x8 c) {
  return __builtin_amdgcn_wmma_f32_16x16x32_bf16(false, a, false, b, (short)0, c, false, false);
}

static __device__ __forceinline__ f32x8 zero8() {
  f32x8 z;
#pragma unroll
  for (int e = 0; e < 8; ++e) z[e] = 0.0f;
  return z;
}

static __device__ __forceinline__ bf16 to_bf16(float f) { return (bf16)f; }

static __device__ __forceinline__ float bfpair_sum(unsigned u) {
  return __uint_as_float(u << 16) + __uint_as_float(u & 0xffff0000u);
}

// Async DMA: global -> LDS, 16 bytes per lane, tracked by ASYNCcnt (CDNA5).
// LDS offset = low 32 bits of the generic pointer (flat aperture rule).
static __device__ __forceinline__ void async_b128(bf16* lds_dst, const bf16* gsrc) {
  unsigned off = (unsigned)(size_t)lds_dst;
  asm volatile("global_load_async_to_lds_b128 %0, %1, off"
               :: "v"(off), "v"(gsrc) : "memory");
}
static __device__ __forceinline__ void wait_async0() {
  asm volatile("s_wait_asynccnt 0" ::: "memory");
}

// A-matrix 16x32 bf16 fragment (ISA 7.12.2): lane<16 -> row=lane, K {0..7}+{16..23};
// lane>=16 -> row=lane-16, K {8..15}+{24..31}. Two 16B chunks per lane.
static __device__ __forceinline__ bf16x16 load_a_frag(const bf16* tile, int lane, int ldk) {
  const int row = lane & 15;
  const int k0  = (lane >> 4) << 3;            // 0 or 8
  const bf16* p = tile + row * ldk;
  bf16x16 a;
  uint4* av = reinterpret_cast<uint4*>(&a);
  av[0] = *reinterpret_cast<const uint4*>(p + k0);
  av[1] = *reinterpret_cast<const uint4*>(p + k0 + 16);
  return a;
}

// B-matrix 32x16 bf16 fragment from a column-major (B^T) tile:
// lane<16 -> col=lane, K {0..7}+{16..23}; lane>=16 -> col=lane-16, K {8..15}+{24..31}.
static __device__ __forceinline__ bf16x16 load_b_frag(const bf16* tileT, int lane, int ldk) {
  const int col = lane & 15;
  const int k0  = (lane >> 4) << 3;            // 0 or 8
  const bf16* p = tileT + col * ldk + k0;
  bf16x16 b;
  uint4* bv = reinterpret_cast<uint4*>(&b);
  bv[0] = *reinterpret_cast<const uint4*>(p);
  bv[1] = *reinterpret_cast<const uint4*>(p + 16);
  return b;
}

// ---------------- Stage 1: fp32 -> bf16 cast ----------------
__global__ void cast_f32_to_bf16(const float* __restrict__ s, bf16* __restrict__ d, int n) {
  int i = (blockIdx.x * 256 + threadIdx.x) * 4;
  if (i + 4 <= n) {
    float4 v = *reinterpret_cast<const float4*>(s + i);
    d[i + 0] = to_bf16(v.x);
    d[i + 1] = to_bf16(v.y);
    d[i + 2] = to_bf16(v.z);
    d[i + 3] = to_bf16(v.w);
  }
}

// ---------------- Stage 2: qkv = x @ Wqkv^T, relu(q,k), scatter q /(kT,vT) -----
// block tile 128(M) x 128(cols), BK=32, 8 waves (2M x 4N), wave tile 64x32.
__global__ __launch_bounds__(256) void gemm_qkv(
    const bf16* __restrict__ X,     // (16384, 512)
    const bf16* __restrict__ W,     // (1536, 512) row n = output column n
    bf16* __restrict__ Qb,          // (BH, 4096, 64)
    bf16* __restrict__ KTb,         // (BH, 64, 4096)
    bf16* __restrict__ VTb)         // (BH, 64, 4096)
{
  __shared__ __align__(16) bf16 At[2][128 * 32];
  __shared__ __align__(16) bf16 Bt[2][128 * 32];
  const int tid  = threadIdx.x;
  const int lane = tid & 31;
  const int wave = tid >> 5;
  const int wm   = wave & 1;        // 0..1
  const int wn   = wave >> 1;       // 0..3
  const int m0   = blockIdx.x * 128;
  const int n0   = blockIdx.y * 128;

  f32x8 acc[4][2];
#pragma unroll
  for (int mt = 0; mt < 4; ++mt)
#pragma unroll
    for (int nt = 0; nt < 2; ++nt) acc[mt][nt] = zero8();

  auto load_tiles = [&](int k0, int buf) {
    const int r = tid >> 1;
    const int c = (tid & 1) << 4;
    async_b128(&At[buf][r * 32 + c],     X + (m0 + r) * CDIM + k0 + c);
    async_b128(&At[buf][r * 32 + c + 8], X + (m0 + r) * CDIM + k0 + c + 8);
    async_b128(&Bt[buf][r * 32 + c],     W + (n0 + r) * CDIM + k0 + c);
    async_b128(&Bt[buf][r * 32 + c + 8], W + (n0 + r) * CDIM + k0 + c + 8);
  };

  load_tiles(0, 0);
  wait_async0();
  __syncthreads();
#pragma unroll 1
  for (int kk = 0; kk < CDIM / 32; ++kk) {
    const int buf = kk & 1;
    if (kk + 1 < CDIM / 32) load_tiles((kk + 1) * 32, buf ^ 1);
    if (kk + 2 < CDIM / 32) {
      __builtin_prefetch(X + (m0 + (tid >> 1)) * CDIM + (kk + 2) * 32, 0, 1);
    }
    bf16x16 af[4], bfr[2];
#pragma unroll
    for (int mt = 0; mt < 4; ++mt)
      af[mt] = load_a_frag(&At[buf][(wm * 64 + mt * 16) * 32], lane, 32);
#pragma unroll
    for (int nt = 0; nt < 2; ++nt)
      bfr[nt] = load_b_frag(&Bt[buf][(wn * 32 + nt * 16) * 32], lane, 32);
#pragma unroll
    for (int mt = 0; mt < 4; ++mt)
#pragma unroll
      for (int nt = 0; nt < 2; ++nt)
        acc[mt][nt] = wmma_bf(af[mt], bfr[nt], acc[mt][nt]);
    wait_async0();
    __syncthreads();
  }

  // epilogue: scatter with relu on q/k, k & v stored transposed per head
#pragma unroll
  for (int mt = 0; mt < 4; ++mt) {
#pragma unroll
    for (int nt = 0; nt < 2; ++nt) {
      const int col   = n0 + wn * 32 + nt * 16 + (lane & 15);
      const int rbase = m0 + wm * 64 + mt * 16 + ((lane >> 4) << 3);
      const int t   = col >> 9;        // 0=q 1=k 2=v
      const int rem = col & 511;
      const int h   = rem >> 6;
      const int d   = rem & 63;
#pragma unroll
      for (int r = 0; r < 8; ++r) {
        const int m  = rbase + r;          // = b*4096 + n
        const int b  = m >> 12;
        const int n  = m & 4095;
        const int bh = b * H_ + h;
        float v = acc[mt][nt][r];
        if (t == 0) {
          v = fmaxf(v, 0.0f);
          Qb[((size_t)bh * SEQ + n) * HD + d] = to_bf16(v);
        } else if (t == 1) {
          v = fmaxf(v, 0.0f);
          KTb[((size_t)bh * HD + d) * SEQ + n] = to_bf16(v);
        } else {
          VTb[((size_t)bh * HD + d) * SEQ + n] = to_bf16(v);
        }
      }
    }
  }
}

// ---------------- Stage 3: per (b,h): attn = (k^T v)*temp (64x64), ksum ------
// 8 waves: wm = wave&3 (16 rows each), wn = wave>>2 (2 x 16-col tiles each). K=4096.
__global__ __launch_bounds__(256) void kv_head(
    const bf16* __restrict__ KTb, const bf16* __restrict__ VTb,
    const float* __restrict__ temp,
    bf16* __restrict__ AttnT,      // (BH, 64(e), 64(d))  transposed for stage 4 B-frags
    float* __restrict__ Ksum)      // (BH, 64)
{
  __shared__ __align__(16) bf16 Kt[2][64 * 32];
  __shared__ __align__(16) bf16 Vt[2][64 * 32];
  __shared__ float red[64][4];
  const int bh = blockIdx.x;
  const int h  = bh & (H_ - 1);
  const bf16* KT = KTb + (size_t)bh * HD * SEQ;
  const bf16* VT = VTb + (size_t)bh * HD * SEQ;
  const int tid  = threadIdx.x;
  const int lane = tid & 31;
  const int wave = tid >> 5;
  const int wm   = wave & 3;
  const int wn   = wave >> 2;

  f32x8 acc[2];
  acc[0] = zero8(); acc[1] = zero8();

  auto load_tiles = [&](int k0, int buf) {
    const int tsel = tid >> 7;            // 0 = K tile, 1 = V tile
    const int idx  = tid & 127;
    const int r    = idx >> 1;            // 0..63
    const int c    = (idx & 1) << 4;
    const bf16* src = (tsel ? VT : KT) + r * SEQ + k0 + c;
    bf16* dst = (tsel ? &Vt[buf][r * 32 + c] : &Kt[buf][r * 32 + c]);
    async_b128(dst, src);
    async_b128(dst + 8, src + 8);
  };

  load_tiles(0, 0);
  wait_async0();
  __syncthreads();
#pragma unroll 1
  for (int kk = 0; kk < SEQ / 32; ++kk) {
    const int buf = kk & 1;
    if (kk + 1 < SEQ / 32) load_tiles((kk + 1) * 32, buf ^ 1);
    bf16x16 af = load_a_frag(&Kt[buf][(wm * 16) * 32], lane, 32);
#pragma unroll
    for (int nt = 0; nt < 2; ++nt) {
      bf16x16 bfr = load_b_frag(&Vt[buf][(wn * 32 + nt * 16) * 32], lane, 32);
      acc[nt] = wmma_bf(af, bfr, acc[nt]);
    }
    wait_async0();
    __syncthreads();
  }

  const float tscale = temp[h];
#pragma unroll
  for (int nt = 0; nt < 2; ++nt) {
    const int e = wn * 32 + nt * 16 + (lane & 15);
    const int dbase = wm * 16 + ((lane >> 4) << 3);
#pragma unroll
    for (int r = 0; r < 8; ++r) {
      AttnT[(size_t)bh * HD * HD + e * HD + (dbase + r)] = to_bf16(acc[nt][r] * tscale);
    }
  }

  // ksum[d] = sum_n k[n][d] = row sums of KT
  const int row = tid >> 2;
  const int q4  = tid & 3;
  float s = 0.0f;
  const uint4* p = reinterpret_cast<const uint4*>(KT + row * SEQ + q4 * 1024);
#pragma unroll 4
  for (int j = 0; j < 128; ++j) {
    uint4 u = p[j];
    s += bfpair_sum(u.x) + bfpair_sum(u.y) + bfpair_sum(u.z) + bfpair_sum(u.w);
  }
  red[row][q4] = s;
  __syncthreads();
  if (q4 == 0)
    Ksum[bh * HD + row] = red[row][0] + red[row][1] + red[row][2] + red[row][3];
}

// ---------------- Stage 3.5: denom = clamp(q . ksum, 100) --------------------
__global__ void denom_kernel(const bf16* __restrict__ Qb, const float* __restrict__ Ksum,
                             float* __restrict__ Denom) {
  const int idx = blockIdx.x * 256 + threadIdx.x;      // BH*SEQ
  if (idx >= BH * SEQ) return;
  const int bh = idx >> 12;
  const unsigned* q = reinterpret_cast<const unsigned*>(Qb + (size_t)idx * HD);
  const float* ks = Ksum + bh * HD;
  float s = 0.0f;
#pragma unroll
  for (int d2 = 0; d2 < 32; ++d2) {
    unsigned u = q[d2];
    s += __uint_as_float(u << 16) * ks[2 * d2] +
         __uint_as_float(u & 0xffff0000u) * ks[2 * d2 + 1];
  }
  Denom[idx] = fmaxf(s, 100.0f);
}

// ---------------- Stage 4: attn1 = q @ attn ; out = attn1/denom -------------
// per (bh): block tile 128(n) x 64(e), K=64. 8 waves: wm=wave&3 (32 rows), wn=wave>>2 (32 cols).
__global__ __launch_bounds__(256) void attn1_kernel(
    const bf16* __restrict__ Qb, const bf16* __restrict__ AttnT,
    const float* __restrict__ Denom,
    float* __restrict__ A1,        // d_out part 2: (BH, 4096, 64) fp32
    bf16* __restrict__ OutB)       // (B*4096, 512) bf16 for projection
{
  __shared__ __align__(16) bf16 Qt[2][128 * 32];
  __shared__ __align__(16) bf16 Bt[64 * 64];
  const int tid  = threadIdx.x;
  const int lane = tid & 31;
  const int wave = tid >> 5;
  const int wm   = wave & 3;
  const int wn   = wave >> 2;
  const int bh   = blockIdx.y;
  const int b    = bh >> 3;
  const int h    = bh & 7;
  const int m0   = blockIdx.x * 128;
  const bf16* Q  = Qb + (size_t)bh * SEQ * HD;

  { // async-load full 64x64 attn^T tile (8 KB)
    const bf16* src = AttnT + (size_t)bh * HD * HD + tid * 16;
    async_b128(&Bt[tid * 16], src);
    async_b128(&Bt[tid * 16 + 8], src + 8);
  }
  auto load_q = [&](int k0, int buf) {
    const int r = tid >> 1;
    const int c = (tid & 1) << 4;
    const bf16* g = Q + (m0 + r) * HD + k0 + c;
    async_b128(&Qt[buf][r * 32 + c], g);
    async_b128(&Qt[buf][r * 32 + c + 8], g + 8);
  };
  load_q(0, 0);
  wait_async0();
  __syncthreads();

  f32x8 acc[2][2];
#pragma unroll
  for (int mt = 0; mt < 2; ++mt)
#pragma unroll
    for (int nt = 0; nt < 2; ++nt) acc[mt][nt] = zero8();

#pragma unroll
  for (int kk = 0; kk < 2; ++kk) {
    const int buf = kk;
    if (kk == 0) load_q(32, 1);
    bf16x16 af[2], bfr[2];
#pragma unroll
    for (int mt = 0; mt < 2; ++mt)
      af[mt] = load_a_frag(&Qt[buf][(wm * 32 + mt * 16) * 32], lane, 32);
#pragma unroll
    for (int nt = 0; nt < 2; ++nt)
      bfr[nt] = load_b_frag(&Bt[(wn * 32 + nt * 16) * HD + kk * 32], lane, HD);
#pragma unroll
    for (int mt = 0; mt < 2; ++mt)
#pragma unroll
      for (int nt = 0; nt < 2; ++nt)
        acc[mt][nt] = wmma_bf(af[mt], bfr[nt], acc[mt][nt]);
    wait_async0();
    __syncthreads();
  }

#pragma unroll
  for (int mt = 0; mt < 2; ++mt) {
#pragma unroll
    for (int nt = 0; nt < 2; ++nt) {
      const int e  = wn * 32 + nt * 16 + (lane & 15);
      const int nb = m0 + wm * 32 + mt * 16 + ((lane >> 4) << 3);
#pragma unroll
      for (int r = 0; r < 8; ++r) {
        const int n = nb + r;
        const float a1 = acc[mt][nt][r];
        A1[((size_t)bh * SEQ + n) * HD + e] = a1;
        const float dnm = Denom[bh * SEQ + n];
        OutB[((size_t)(b * SEQ + n)) * CDIM + h * HD + e] = to_bf16(a1 / dnm);
      }
    }
  }
}

// ---------------- Stage 5: y = out @ Wproj^T + b ----------------------------
__global__ __launch_bounds__(256) void gemm_proj(
    const bf16* __restrict__ X,     // (16384, 512)
    const bf16* __restrict__ W,     // (512, 512)
    const float* __restrict__ Bias, // (512)
    float* __restrict__ Y)          // d_out part 1: (16384, 512) fp32
{
  __shared__ __align__(16) bf16 At[2][128 * 32];
  __shared__ __align__(16) bf16 Bt[2][128 * 32];
  const int tid  = threadIdx.x;
  const int lane = tid & 31;
  const int wave = tid >> 5;
  const int wm   = wave & 1;
  const int wn   = wave >> 1;
  const int m0   = blockIdx.x * 128;
  const int n0   = blockIdx.y * 128;

  f32x8 acc[4][2];
#pragma unroll
  for (int mt = 0; mt < 4; ++mt)
#pragma unroll
    for (int nt = 0; nt < 2; ++nt) acc[mt][nt] = zero8();

  auto load_tiles = [&](int k0, int buf) {
    const int r = tid >> 1;
    const int c = (tid & 1) << 4;
    async_b128(&At[buf][r * 32 + c],     X + (m0 + r) * CDIM + k0 + c);
    async_b128(&At[buf][r * 32 + c + 8], X + (m0 + r) * CDIM + k0 + c + 8);
    async_b128(&Bt[buf][r * 32 + c],     W + (n0 + r) * CDIM + k0 + c);
    async_b128(&Bt[buf][r * 32 + c + 8], W + (n0 + r) * CDIM + k0 + c + 8);
  };

  load_tiles(0, 0);
  wait_async0();
  __syncthreads();
#pragma unroll 1
  for (int kk = 0; kk < CDIM / 32; ++kk) {
    const int buf = kk & 1;
    if (kk + 1 < CDIM / 32) load_tiles((kk + 1) * 32, buf ^ 1);
    if (kk + 2 < CDIM / 32) {
      __builtin_prefetch(X + (m0 + (tid >> 1)) * CDIM + (kk + 2) * 32, 0, 1);
    }
    bf16x16 af[4], bfr[2];
#pragma unroll
    for (int mt = 0; mt < 4; ++mt)
      af[mt] = load_a_frag(&At[buf][(wm * 64 + mt * 16) * 32], lane, 32);
#pragma unroll
    for (int nt = 0; nt < 2; ++nt)
      bfr[nt] = load_b_frag(&Bt[buf][(wn * 32 + nt * 16) * 32], lane, 32);
#pragma unroll
    for (int mt = 0; mt < 4; ++mt)
#pragma unroll
      for (int nt = 0; nt < 2; ++nt)
        acc[mt][nt] = wmma_bf(af[mt], bfr[nt], acc[mt][nt]);
    wait_async0();
    __syncthreads();
  }

#pragma unroll
  for (int mt = 0; mt < 4; ++mt) {
#pragma unroll
    for (int nt = 0; nt < 2; ++nt) {
      const int col   = n0 + wn * 32 + nt * 16 + (lane & 15);
      const int rbase = m0 + wm * 64 + mt * 16 + ((lane >> 4) << 3);
      const float bias = Bias[col];
#pragma unroll
      for (int r = 0; r < 8; ++r) {
        Y[(size_t)(rbase + r) * CDIM + col] = acc[mt][nt][r] + bias;
      }
    }
  }
}

// ---------------- host launcher ---------------------------------------------
extern "C" void kernel_launch(void* const* d_in, const int* in_sizes, int n_in,
                              void* d_out, int out_size, void* d_ws, size_t ws_size,
                              hipStream_t stream) {
  (void)in_sizes; (void)n_in; (void)out_size; (void)ws_size;
  const float* x     = (const float*)d_in[0];   // (4,4096,512)
  const float* wqkv  = (const float*)d_in[1];   // (1536,512)
  const float* temp  = (const float*)d_in[2];   // (8,1,1)
  const float* wproj = (const float*)d_in[3];   // (512,512)
  const float* bproj = (const float*)d_in[4];   // (512)

  float* y  = (float*)d_out;                       // 16384*512
  float* a1 = (float*)d_out + (size_t)NTOK * CDIM; // (BH,4096,64)

  char* w = (char*)d_ws;
  bf16* x_bf     = (bf16*)w; w += (size_t)NTOK * CDIM * 2;        // 16 MB
  bf16* wqkv_bf  = (bf16*)w; w += (size_t)3 * CDIM * CDIM * 2;    // 1.5 MB
  bf16* wproj_bf = (bf16*)w; w += (size_t)CDIM * CDIM * 2;        // 0.5 MB
  bf16* q_bf     = (bf16*)w; w += (size_t)BH * SEQ * HD * 2;      // 16 MB
  bf16* kT_bf    = (bf16*)w; w += (size_t)BH * HD * SEQ * 2;      // 16 MB
  bf16* vT_bf    = (bf16*)w; w += (size_t)BH * HD * SEQ * 2;      // 16 MB
  bf16* attnT_bf = (bf16*)w; w += (size_t)BH * HD * HD * 2;       // 256 KB
  float* ksum    = (float*)w; w += (size_t)BH * HD * 4;           // 8 KB
  float* denomp  = (float*)w; w += (size_t)BH * SEQ * 4;          // 512 KB
  bf16* out_bf   = x_bf;  // x is dead after gemm_qkv; reuse its 16 MB

  cast_f32_to_bf16<<<(NTOK * CDIM) / 1024, 256, 0, stream>>>(x, x_bf, NTOK * CDIM);
  cast_f32_to_bf16<<<(3 * CDIM * CDIM) / 1024, 256, 0, stream>>>(wqkv, wqkv_bf, 3 * CDIM * CDIM);
  cast_f32_to_bf16<<<(CDIM * CDIM) / 1024, 256, 0, stream>>>(wproj, wproj_bf, CDIM * CDIM);

  gemm_qkv<<<dim3(NTOK / 128, (3 * CDIM) / 128), 256, 0, stream>>>(
      x_bf, wqkv_bf, q_bf, kT_bf, vT_bf);

  kv_head<<<BH, 256, 0, stream>>>(kT_bf, vT_bf, temp, attnT_bf, ksum);

  denom_kernel<<<(BH * SEQ) / 256, 256, 0, stream>>>(q_bf, ksum, denomp);

  attn1_kernel<<<dim3(SEQ / 128, BH), 256, 0, stream>>>(
      q_bf, attnT_bf, denomp, a1, out_bf);

  gemm_proj<<<dim3(NTOK / 128, CDIM / 128), 256, 0, stream>>>(
      out_bf, wproj_bf, bproj, y);
}
